// ConstructQuarter_82025285419619
// MI455X (gfx1250) — matrix operand order repaired and checked by reading, not compiled, not run on hardware
//
#include <hip/hip_runtime.h>
#include <hip/hip_bf16.h>

// ---------------------------------------------------------------------------
// Problem constants (fully-connected graph => An = (ones + I)/2049)
// ---------------------------------------------------------------------------
#define NN   2048     // nodes
#define DD   128      // feature dim (IN == OUT)
#define KK   5        // sampled nodes
#define ITERS 75
#define INV2049 (1.0f / 2049.0f)

typedef _Float16 half_t;
typedef half_t v16h __attribute__((ext_vector_type(16)));
typedef half_t v8h  __attribute__((ext_vector_type(8)));
typedef float  v8f  __attribute__((ext_vector_type(8)));

union F16x16 { v16h v; v8h h[2]; };
union F16x8  { v8h  v; half_t e[8]; };

// ---------------------------------------------------------------------------
// WMMA fragment loaders (CDNA5 16x16x32 f16, wave32 layouts per ISA 7.12.2)
// A operand: row-major [M][K] f16.  lane m = lane&15
//   lane<16 : K = {k0..k0+7, k0+16..k0+23};  lane>=16 : K = {k0+8.., k0+24..}
// B operand: we store B^T row-major [N][K].  lane n = lane&15
//   lane<16 : K = {k0..k0+7, k0+16..k0+23};  lane>=16 : K = {k0+8.., k0+24..}
// ---------------------------------------------------------------------------
__device__ __forceinline__ v16h load_a_frag(const half_t* A, int lda,
                                            int m0, int k0, int lane) {
    const int m = m0 + (lane & 15);
    const int s = (lane & 16) ? 8 : 0;
    const half_t* p = A + (size_t)m * lda + k0 + s;
    F16x16 u;
    u.h[0] = *(const v8h*)(p);
    u.h[1] = *(const v8h*)(p + 16);
    return u.v;
}

__device__ __forceinline__ v16h load_bt_frag(const half_t* Bt, int ldb,
                                             int n0, int k0, int lane) {
    const int n = n0 + (lane & 15);
    const int t = (lane & 16) ? 8 : 0;
    const half_t* p = Bt + (size_t)n * ldb + k0 + t;
    F16x16 u;
    u.h[0] = *(const v8h*)(p);
    u.h[1] = *(const v8h*)(p + 16);
    return u.v;
}

__device__ __forceinline__ v8f wmma_f16(v16h a, v16h b, v8f c) {
    return __builtin_amdgcn_wmma_f32_16x16x32_f16(false, a, false, b,
                                                  (short)0, c, false, false);
}

__device__ __forceinline__ void atomicMaxF(float* addr, float val) {
    int* ai = (int*)addr;
    int old = *ai;
    while (__int_as_float(old) < val) {
        int assumed = old;
        old = atomicCAS(ai, assumed, __float_as_int(val));
        if (old == assumed) break;
    }
}

// ---------------------------------------------------------------------------
// Small prep kernels
// ---------------------------------------------------------------------------
__global__ void k_init_scalars(float* cmax) {
    if (threadIdx.x == 0) cmax[0] = -1e30f;
}

__global__ void k_f32_to_f16(const float* __restrict__ src,
                             half_t* __restrict__ dst, int n) {
    int i = blockIdx.x * blockDim.x + threadIdx.x;
    if (i < n) dst[i] = (half_t)src[i];
}

// Wt[mat][n][k] = W_mat[k][n]  (transposed f16 copies of the 128x128 weights)
__global__ void k_convert_w(const float* __restrict__ Wg,
                            const float* __restrict__ Wk,
                            const float* __restrict__ Wq,
                            half_t* __restrict__ Wt) {
    int i = blockIdx.x * blockDim.x + threadIdx.x;  // 3*128*128
    if (i >= 3 * DD * DD) return;
    int mat = i / (DD * DD);
    int r   = (i / DD) & (DD - 1);   // n
    int c   = i & (DD - 1);          // k
    const float* W = (mat == 0) ? Wg : (mat == 1) ? Wk : Wq;
    Wt[i] = (half_t)W[c * DD + r];
}

// hT0[d][n] = (f16) init_state[n][d]
__global__ void k_init_h(const float* __restrict__ init_state,
                         half_t* __restrict__ hT) {
    int i = blockIdx.x * blockDim.x + threadIdx.x;  // 128*2048
    if (i >= DD * NN) return;
    int d = i / NN, n = i & (NN - 1);
    hT[(size_t)d * NN + n] = (half_t)init_state[(size_t)n * DD + d];
}

// ---------------------------------------------------------------------------
// Projection GEMM: P[mat] = x_h (2048x128) @ W[mat] (128x128), f32 out.
// grid (128 row-strips, 3 mats), 256 threads = 8 waves, wave w -> cols 16w..
// ---------------------------------------------------------------------------
__global__ void k_proj_gemm(const half_t* __restrict__ xh,
                            const half_t* __restrict__ Wt,
                            float* __restrict__ P) {
    const int lane = threadIdx.x & 31;
    const int wave = threadIdx.x >> 5;
    const int m0 = blockIdx.x * 16;
    const int n0 = wave * 16;
    const int mat = blockIdx.y;
    const half_t* Bt = Wt + (size_t)mat * DD * DD;
    float* Pm = P + (size_t)mat * NN * DD;

    v8f acc = {};
    #pragma unroll
    for (int k0 = 0; k0 < DD; k0 += 32) {
        v16h a = load_a_frag(xh, DD, m0, k0, lane);
        v16h b = load_bt_frag(Bt, DD, n0, k0, lane);
        acc = wmma_f16(a, b, acc);
    }
    const int off = (lane & 16) ? 8 : 0;
    const int col = n0 + (lane & 15);
    #pragma unroll
    for (int j = 0; j < 8; ++j)
        Pm[(size_t)(m0 + off + j) * DD + col] = acc[j];
}

// colsum[mat][d] = sum_r P[mat][r][d]
__global__ void k_colsum(const float* __restrict__ P, float* __restrict__ cs) {
    __shared__ float red[256];
    const int mat = blockIdx.x / DD;
    const int d   = blockIdx.x % DD;
    const float* Pm = P + (size_t)mat * NN * DD;
    float s = 0.f;
    for (int r = threadIdx.x; r < NN; r += 256) s += Pm[(size_t)r * DD + d];
    red[threadIdx.x] = s;
    __syncthreads();
    for (int st = 128; st > 0; st >>= 1) {
        if (threadIdx.x < st) red[threadIdx.x] += red[threadIdx.x + st];
        __syncthreads();
    }
    if (threadIdx.x == 0) cs[mat * DD + d] = red[0];
}

// abstract = (P0 + cs0)/2049 + b_g ;  Kn/Qn = rowl2norm((P{1,2}+cs)/2049 + b)
// one block per row, 128 threads
__global__ void k_transform(const float* __restrict__ P,
                            const float* __restrict__ cs,
                            const float* __restrict__ bg,
                            const float* __restrict__ bk,
                            const float* __restrict__ bq,
                            float*  __restrict__ abstract,
                            half_t* __restrict__ Kn,
                            half_t* __restrict__ Qn) {
    __shared__ float red[DD];
    const int r = blockIdx.x, d = threadIdx.x;
    const size_t idx = (size_t)r * DD + d;

    float g  = (P[idx]                 + cs[d])          * INV2049 + bg[d];
    float kv = (P[(size_t)NN*DD   + idx] + cs[DD + d])   * INV2049 + bk[d];
    float qv = (P[(size_t)2*NN*DD + idx] + cs[2*DD + d]) * INV2049 + bq[d];
    abstract[idx] = g;

    red[d] = kv * kv; __syncthreads();
    for (int st = 64; st > 0; st >>= 1) {
        if (d < st) red[d] += red[d + st];
        __syncthreads();
    }
    float kn = red[0]; __syncthreads();
    red[d] = qv * qv; __syncthreads();
    for (int st = 64; st > 0; st >>= 1) {
        if (d < st) red[d] += red[d + st];
        __syncthreads();
    }
    float qn = red[0];

    Kn[idx] = (half_t)(kv / fmaxf(sqrtf(kn), 1e-8f));
    Qn[idx] = (half_t)(qv / fmaxf(sqrtf(qn), 1e-8f));
}

// ---------------------------------------------------------------------------
// Cosine GEMM: C = Kn (2048x128) @ Qn^T -> f32 [2048][2048]; track global max.
// grid (16, 128): n0 = blockIdx.x*128 + wave*16, m0 = blockIdx.y*16
// ---------------------------------------------------------------------------
__global__ void k_cos_gemm(const half_t* __restrict__ Kn,
                           const half_t* __restrict__ Qn,
                           float* __restrict__ C, float* __restrict__ cmax) {
    __shared__ float wmax[8];
    const int lane = threadIdx.x & 31;
    const int wave = threadIdx.x >> 5;
    const int n0 = blockIdx.x * 128 + wave * 16;
    const int m0 = blockIdx.y * 16;

    v8f acc = {};
    #pragma unroll
    for (int k0 = 0; k0 < DD; k0 += 32) {
        v16h a = load_a_frag(Kn, DD, m0, k0, lane);
        v16h b = load_bt_frag(Qn, DD, n0, k0, lane);
        acc = wmma_f16(a, b, acc);
    }
    const int off = (lane & 16) ? 8 : 0;
    const int col = n0 + (lane & 15);
    float lm = -1e30f;
    #pragma unroll
    for (int j = 0; j < 8; ++j) {
        C[(size_t)(m0 + off + j) * NN + col] = acc[j];
        lm = fmaxf(lm, acc[j]);
    }
    #pragma unroll
    for (int m = 1; m < 32; m <<= 1)
        lm = fmaxf(lm, __shfl_xor(lm, m, 32));
    if (lane == 0) wmax[wave] = lm;
    __syncthreads();
    if (threadIdx.x == 0) {
        float bm = wmax[0];
        #pragma unroll
        for (int w = 1; w < 8; ++w) bm = fmaxf(bm, wmax[w]);
        atomicMaxF(cmax, bm);
    }
}

// Aw = exp(C - max) thresholded at 0.5, stored f16
__global__ void k_build_aw(const float* __restrict__ C,
                           const float* __restrict__ cmax,
                           half_t* __restrict__ Aw) {
    int i = blockIdx.x * blockDim.x + threadIdx.x;
    if (i >= NN * NN) return;
    float w = __expf(C[i] - cmax[0]);
    Aw[i] = (w > 0.5f) ? (half_t)w : (half_t)0.0f;
}

// ---------------------------------------------------------------------------
// Fused propagation step: hT_out = rownorm(Aw @ h)^T  (all f16 in, f16 out)
// grid 128 (16-row strips), 256 threads = 8 waves; wave w -> cols 16w..16w+15
// ---------------------------------------------------------------------------
__global__ void k_prop(const half_t* __restrict__ Aw,
                       const half_t* __restrict__ hT_in,
                       half_t* __restrict__ hT_out) {
    __shared__ float rowsq[16];
    const int lane = threadIdx.x & 31;
    const int wave = threadIdx.x >> 5;
    const int m0 = blockIdx.x * 16;
    const int n0 = wave * 16;
    const int s = (lane & 16) ? 8 : 0;
    const half_t* arow = Aw + (size_t)(m0 + (lane & 15)) * NN + s;

    if (threadIdx.x < 16) rowsq[threadIdx.x] = 0.f;
    __syncthreads();

    v8f acc = {};
    for (int k0 = 0; k0 < NN; k0 += 32) {
        if (k0 + 32 < NN)
            __builtin_prefetch(arow + k0 + 32, 0, 1);   // global_prefetch_b8
        v16h a = load_a_frag(Aw, NN, m0, k0, lane);
        v16h b = load_bt_frag(hT_in, NN, n0, k0, lane);
        acc = wmma_f16(a, b, acc);
    }

    // per-row sum of squares: reduce across the 16 lanes of each half-wave
    v8f sq;
    #pragma unroll
    for (int j = 0; j < 8; ++j) sq[j] = acc[j] * acc[j];
    #pragma unroll
    for (int m = 1; m < 16; m <<= 1) {
        #pragma unroll
        for (int j = 0; j < 8; ++j) sq[j] += __shfl_xor(sq[j], m, 32);
    }
    const int off = (lane & 16) ? 8 : 0;
    if ((lane & 15) == 0) {
        #pragma unroll
        for (int j = 0; j < 8; ++j) atomicAdd(&rowsq[off + j], sq[j]);
    }
    __syncthreads();

    // scale, pack to f16, one 16B store per lane into transposed layout
    const int col = n0 + (lane & 15);
    F16x8 out;
    #pragma unroll
    for (int j = 0; j < 8; ++j) {
        float sc = 1.0f / fmaxf(sqrtf(rowsq[off + j]), 1e-12f);
        out.e[j] = (half_t)(acc[j] * sc);
    }
    *(v8h*)(hT_out + (size_t)col * NN + m0 + off) = out.v;
}

// ---------------------------------------------------------------------------
// sim[k][n] = <h[sample_k], h[n]>  (rows of h are already unit-norm)
// ---------------------------------------------------------------------------
__global__ void k_sim(const half_t* __restrict__ hT,
                      const int* __restrict__ sample,
                      float* __restrict__ sim) {
    __shared__ float hs[KK][DD];
    for (int idx = threadIdx.x; idx < KK * DD; idx += 256) {
        int k = idx / DD, d = idx % DD;
        hs[k][d] = (float)hT[(size_t)d * NN + sample[k]];
    }
    __syncthreads();
    int n = blockIdx.x * 256 + threadIdx.x;
    float a[KK] = {0.f, 0.f, 0.f, 0.f, 0.f};
    for (int d = 0; d < DD; ++d) {
        float v = (float)hT[(size_t)d * NN + n];
        #pragma unroll
        for (int k = 0; k < KK; ++k) a[k] += v * hs[k][d];
    }
    #pragma unroll
    for (int k = 0; k < KK; ++k) sim[(size_t)k * NN + n] = a[k];
}

__global__ void k_rowmax(const float* __restrict__ sim,
                         float* __restrict__ rowmax,
                         float* __restrict__ scores_out) {
    __shared__ float red[256];
    const int k = blockIdx.x;
    float m = -1e30f;
    for (int n = threadIdx.x; n < NN; n += 256)
        m = fmaxf(m, sim[(size_t)k * NN + n]);
    red[threadIdx.x] = m; __syncthreads();
    for (int st = 128; st > 0; st >>= 1) {
        if (threadIdx.x < st)
            red[threadIdx.x] = fmaxf(red[threadIdx.x], red[threadIdx.x + st]);
        __syncthreads();
    }
    if (threadIdx.x == 0) {
        rowmax[k] = red[0];
        scores_out[k] = 1.0f;   // max of exp(x - max(x)) is exactly 1
    }
}

__global__ void k_masks(const float* __restrict__ sim,
                        const float* __restrict__ rowmax,
                        float* __restrict__ masks_out) {
    int i = blockIdx.x * blockDim.x + threadIdx.x;
    if (i >= KK * NN) return;
    int k = i / NN;
    masks_out[i] = __expf(sim[i] - rowmax[k]);
}

// node_features[k][d] = sum_n masks[k][n] * abstract[n][d]
__global__ void k_nodefeat(const float* __restrict__ masks,
                           const float* __restrict__ abstract,
                           float* __restrict__ nf_out) {
    const int k = blockIdx.x, d = threadIdx.x;
    float s = 0.f;
    for (int n = 0; n < NN; ++n)
        s += masks[(size_t)k * NN + n] * abstract[(size_t)n * DD + d];
    nf_out[(size_t)k * DD + d] = s;
}

// ---------------------------------------------------------------------------
// Host-side launch
// ---------------------------------------------------------------------------
extern "C" void kernel_launch(void* const* d_in, const int* in_sizes, int n_in,
                              void* d_out, int out_size, void* d_ws, size_t ws_size,
                              hipStream_t stream) {
    const float* x        = (const float*)d_in[0];
    // d_in[1] = edge_index (fully connected by construction; unused)
    const float* initst   = (const float*)d_in[2];
    const float* Wg       = (const float*)d_in[3];
    const float* bg       = (const float*)d_in[4];
    const float* Wk       = (const float*)d_in[5];
    const float* bk       = (const float*)d_in[6];
    const float* Wq       = (const float*)d_in[7];
    const float* bq       = (const float*)d_in[8];
    const int*   sample   = (const int*)d_in[9];
    float* out = (float*)d_out;

    char* base = (char*)d_ws;
    size_t off = 0;
    auto take = [&](size_t bytes) -> void* {
        void* p = base + off;
        off += (bytes + 255) & ~(size_t)255;
        return p;
    };
    float*  C        = (float*) take((size_t)NN * NN * 4);   // 16 MiB
    half_t* Aw       = (half_t*)take((size_t)NN * NN * 2);   //  8 MiB
    half_t* xh       = (half_t*)take((size_t)NN * DD * 2);
    half_t* Wt       = (half_t*)take((size_t)3 * DD * DD * 2);
    float*  P        = (float*) take((size_t)3 * NN * DD * 4);
    float*  cs       = (float*) take((size_t)3 * DD * 4);
    float*  abstract = (float*) take((size_t)NN * DD * 4);
    half_t* Kn       = (half_t*)take((size_t)NN * DD * 2);
    half_t* Qn       = (half_t*)take((size_t)NN * DD * 2);
    half_t* hTa      = (half_t*)take((size_t)DD * NN * 2);
    half_t* hTb      = (half_t*)take((size_t)DD * NN * 2);
    float*  sim      = (float*) take((size_t)KK * NN * 4);
    float*  rowmax   = (float*) take(64);
    float*  cmax     = (float*) take(64);

    float* nf_out     = out;                 // [5,128]
    float* masks_out  = out + KK * DD;       // [5,2048]
    float* scores_out = out + KK * DD + KK * NN;  // [5]

    // --- prep ---
    k_init_scalars<<<1, 32, 0, stream>>>(cmax);
    k_f32_to_f16<<<(NN * DD + 255) / 256, 256, 0, stream>>>(x, xh, NN * DD);
    k_convert_w<<<(3 * DD * DD + 255) / 256, 256, 0, stream>>>(Wg, Wk, Wq, Wt);
    k_init_h<<<(DD * NN + 255) / 256, 256, 0, stream>>>(initst, hTa);

    // --- projections + GCN transform ---
    k_proj_gemm<<<dim3(NN / 16, 3), 256, 0, stream>>>(xh, Wt, P);
    k_colsum<<<3 * DD, 256, 0, stream>>>(P, cs);
    k_transform<<<NN, DD, 0, stream>>>(P, cs, bg, bk, bq, abstract, Kn, Qn);

    // --- cosine similarity matrix + edge weights ---
    k_cos_gemm<<<dim3(NN / 128, NN / 16), 256, 0, stream>>>(Kn, Qn, C, cmax);
    k_build_aw<<<(NN * NN + 255) / 256, 256, 0, stream>>>(C, cmax, Aw);

    // --- 75 propagation iterations ---
    half_t* hin = hTa;
    half_t* hout = hTb;
    for (int it = 0; it < ITERS; ++it) {
        k_prop<<<NN / 16, 256, 0, stream>>>(Aw, hin, hout);
        half_t* t = hin; hin = hout; hout = t;
    }
    half_t* hfin = hin;  // result of last iteration

    // --- node extraction ---
    k_sim<<<NN / 256, 256, 0, stream>>>(hfin, sample, sim);
    k_rowmax<<<KK, 256, 0, stream>>>(sim, rowmax, scores_out);
    k_masks<<<(KK * NN + 255) / 256, 256, 0, stream>>>(sim, rowmax, masks_out);
    k_nodefeat<<<KK, DD, 0, stream>>>(masks_out, abstract, nf_out);
}
